// BinarizeLinear_71416716198470
// MI455X (gfx1250) — matrix operand order repaired and checked by reading, not compiled, not run on hardware
//
#include <hip/hip_runtime.h>

typedef __bf16 bf16_t;
typedef __attribute__((ext_vector_type(4)))  __bf16 v4bf;
typedef __attribute__((ext_vector_type(8)))  __bf16 v8bf;
typedef __attribute__((ext_vector_type(16))) __bf16 v16bf;
typedef __attribute__((ext_vector_type(8)))  float  v8f;

#define BDIM 32768   // batch (M)
#define KDIM 1024    // reduction
#define NDIM 512     // units (N)
#define BN_EPS 1e-3f

__device__ __forceinline__ float signf(float v) {
    return (v > 0.0f) ? 1.0f : ((v < 0.0f) ? -1.0f : 0.0f);
}

// ---------------------------------------------------------------------------
// Kernel 0: zero the column sum / sumsq accumulators (2*NDIM floats in ws)
// ---------------------------------------------------------------------------
__global__ void zero_ws_kernel(float* __restrict__ ws) {
    int i = blockIdx.x * blockDim.x + threadIdx.x;   // grid covers exactly 2*NDIM
    ws[i] = 0.0f;
}

// ---------------------------------------------------------------------------
// Kernel 1: binarize W once: Wb^T[n][k] = (bf16)sign(W[k][n])
//   Pre-transposed to N-major so the GEMM's B-fragment staging is a straight
//   contiguous bf16 copy (no per-tile sign math, half the W bytes).
// ---------------------------------------------------------------------------
__global__ void binarize_w_kernel(const float* __restrict__ W,
                                  bf16_t* __restrict__ WbT) {
    int i = blockIdx.x * blockDim.x + threadIdx.x;   // 0 .. KDIM*NDIM-1
    int k = i >> 9;            // / NDIM
    int n = i & (NDIM - 1);
    WbT[(size_t)n * KDIM + k] = (bf16_t)signf(W[i]);
}

// ---------------------------------------------------------------------------
// Kernel 2: WMMA GEMM  out = x @ sign(W)   (f32 via split-bf16 hi/lo)
//   block tile: 128 (M) x 128 (N), 256 threads = 8 waves,
//   wave tile 32x64 = 2x4 WMMA tiles -> 16 v_wmma per K-step.
//   Writes f32 pre-activation into `out` and accumulates per-column
//   sum / sum-of-squares with global f32 atomics.
// ---------------------------------------------------------------------------
__launch_bounds__(256, 2)
__global__ void bgemm_wmma_kernel(const float* __restrict__ X,
                                  const bf16_t* __restrict__ WbT,
                                  float* __restrict__ out,
                                  float* __restrict__ colsum,
                                  float* __restrict__ colsq) {
    __shared__ __align__(32) bf16_t xhi[128][32];
    __shared__ __align__(32) bf16_t xlo[128][32];
    __shared__ __align__(32) bf16_t wbT[128][32];   // n-major tile of Wb^T

    const int tid  = threadIdx.x;
    const int lane = tid & 31;
    const int wave = tid >> 5;        // 0..7
    const int wm   = wave >> 1;       // 0..3  (M direction, 32 rows each)
    const int wn   = wave & 1;        // 0..1  (N direction, 64 cols each)
    const int m0   = blockIdx.y * 128;
    const int n0   = blockIdx.x * 128;

    v8f acc[2][4] = {};               // 2(M) x 4(N) tiles of 16x16 f32

    // A-fragment addressing (16x32 bf16): lanes 0-15 -> K {0-7,16-23},
    // lanes 16-31 -> K {8-15,24-31}; row = lane&15.
    const int rowA  = wm * 32 + (lane & 15);
    const int kbase = (lane >> 4) * 8;
    // B-fragment addressing (32x16 bf16): lane holds one column N = lane&15,
    // lanes 0-15 -> K {0-7,16-23}, lanes 16-31 -> K {8-15,24-31} (matches A).
    const int nB    = wn * 64 + (lane & 15);
    const int khalf = (lane >> 4) * 8;

    for (int k0 = 0; k0 < KDIM; k0 += 32) {
        // ---- stage x tile: 128 rows x 32 K of f32 -> hi/lo bf16 ----
        #pragma unroll
        for (int i = 0; i < 4; ++i) {
            int c  = tid + 256 * i;            // 0..1023 float4 chunks
            int r  = c >> 3;                   // 0..127
            int cv = (c & 7) * 4;              // K offset within tile
            const float4 v =
                *(const float4*)(X + (size_t)(m0 + r) * KDIM + k0 + cv);
            v4bf hi, lo;
            hi.x = (bf16_t)v.x; lo.x = (bf16_t)(v.x - (float)hi.x);
            hi.y = (bf16_t)v.y; lo.y = (bf16_t)(v.y - (float)hi.y);
            hi.z = (bf16_t)v.z; lo.z = (bf16_t)(v.z - (float)hi.z);
            hi.w = (bf16_t)v.w; lo.w = (bf16_t)(v.w - (float)hi.w);
            *(v4bf*)&xhi[r][cv] = hi;
            *(v4bf*)&xlo[r][cv] = lo;
        }
        // ---- stage Wb^T tile: 128 n-rows x 32 K of bf16, straight copy ----
        {
            int nrow = tid >> 1;               // 0..127
            int koff = (tid & 1) * 16;         // 0 or 16
            const bf16_t* src = WbT + (size_t)(n0 + nrow) * KDIM + k0 + koff;
            *(v8bf*)&wbT[nrow][koff + 0] = *(const v8bf*)(src + 0);
            *(v8bf*)&wbT[nrow][koff + 8] = *(const v8bf*)(src + 8);
        }
        // prefetch next K-step tiles into cache (global_prefetch_b8)
        if (k0 + 32 < KDIM) {
            __builtin_prefetch(
                X + (size_t)(m0 + (tid >> 1)) * KDIM + (k0 + 32) + (tid & 1) * 16, 0, 3);
            __builtin_prefetch(
                WbT + (size_t)(n0 + (tid >> 1)) * KDIM + (k0 + 32) + (tid & 1) * 16, 0, 3);
        }
        __syncthreads();

        // ---- 2x4 WMMA tiles, hi+lo contributions (16 wmma / K-step) ----
        v16bf bfr[4];
        #pragma unroll
        for (int tn = 0; tn < 4; ++tn)
            bfr[tn] = __builtin_shufflevector(
                *(const v8bf*)&wbT[nB + tn * 16][khalf],
                *(const v8bf*)&wbT[nB + tn * 16][16 + khalf],
                0,1,2,3,4,5,6,7,8,9,10,11,12,13,14,15);
        #pragma unroll
        for (int tm = 0; tm < 2; ++tm) {
            const bf16_t* ah = &xhi[rowA + tm * 16][0];
            v8bf h0 = *(const v8bf*)(ah + kbase);
            v8bf h1 = *(const v8bf*)(ah + 16 + kbase);
            v16bf a_hi = __builtin_shufflevector(
                h0, h1, 0,1,2,3,4,5,6,7,8,9,10,11,12,13,14,15);
            const bf16_t* al = &xlo[rowA + tm * 16][0];
            v8bf l0 = *(const v8bf*)(al + kbase);
            v8bf l1 = *(const v8bf*)(al + 16 + kbase);
            v16bf a_lo = __builtin_shufflevector(
                l0, l1, 0,1,2,3,4,5,6,7,8,9,10,11,12,13,14,15);
            #pragma unroll
            for (int tn = 0; tn < 4; ++tn) {
                acc[tm][tn] = __builtin_amdgcn_wmma_f32_16x16x32_bf16(
                    false, a_lo, false, bfr[tn], (short)0, acc[tm][tn], false, false);
                acc[tm][tn] = __builtin_amdgcn_wmma_f32_16x16x32_bf16(
                    false, a_hi, false, bfr[tn], (short)0, acc[tm][tn], false, false);
            }
        }
        __syncthreads();
    }

    // ---- writeback + batch-stat accumulation ----
    // C/D layout: VGPR r, lanes 0-15 -> M=r, N=lane; lanes 16-31 -> M=r+8.
    const int mrow = m0 + wm * 32 + (lane >> 4) * 8;
    const int ncol = n0 + wn * 64 + (lane & 15);
    #pragma unroll
    for (int tn = 0; tn < 4; ++tn) {
        const int n = ncol + tn * 16;
        float s = 0.0f, q = 0.0f;
        #pragma unroll
        for (int tm = 0; tm < 2; ++tm) {
            #pragma unroll
            for (int r = 0; r < 8; ++r) {
                float v = acc[tm][tn][r];
                out[(size_t)(mrow + tm * 16 + r) * NDIM + n] = v;
                s += v;
                q += v * v;
            }
        }
        atomicAdd(&colsum[n], s);
        atomicAdd(&colsq[n], q);
    }
}

// ---------------------------------------------------------------------------
// Kernel 3: fold BN into per-column scale/shift
//   s = gamma * rsqrt(var + eps);  t = beta - mean * s
// ---------------------------------------------------------------------------
__global__ void bn_stats_kernel(const float* __restrict__ colsum,
                                const float* __restrict__ colsq,
                                const float* __restrict__ gamma,
                                const float* __restrict__ beta,
                                float* __restrict__ scl,
                                float* __restrict__ sft) {
    int n = blockIdx.x * blockDim.x + threadIdx.x;
    if (n < NDIM) {
        const float inv_b = 1.0f / (float)BDIM;
        float mean = colsum[n] * inv_b;
        float var  = colsq[n] * inv_b - mean * mean;
        float s    = gamma[n] * rsqrtf(var + BN_EPS);
        scl[n] = s;
        sft[n] = beta[n] - mean * s;
    }
}

// ---------------------------------------------------------------------------
// Kernel 4: in-place  out = sign(out * s[n] + t[n]),  float4 vectorized
// ---------------------------------------------------------------------------
__global__ void bn_sign_kernel(float* __restrict__ out,
                               const float* __restrict__ scl,
                               const float* __restrict__ sft) {
    size_t i4 = (size_t)blockIdx.x * blockDim.x + threadIdx.x;
    float4* p = (float4*)out + i4;
    int n = (int)((i4 * 4) & (NDIM - 1));   // NDIM is a power of two, /4 aligned
    float4 v = *p;
    v.x = signf(v.x * scl[n + 0] + sft[n + 0]);
    v.y = signf(v.y * scl[n + 1] + sft[n + 1]);
    v.z = signf(v.z * scl[n + 2] + sft[n + 2]);
    v.w = signf(v.w * scl[n + 3] + sft[n + 3]);
    *p = v;
}

// ---------------------------------------------------------------------------
extern "C" void kernel_launch(void* const* d_in, const int* in_sizes, int n_in,
                              void* d_out, int out_size, void* d_ws, size_t ws_size,
                              hipStream_t stream) {
    const float* x     = (const float*)d_in[0];   // [32768,1024]
    const float* W     = (const float*)d_in[1];   // [1024,512]
    const float* gamma = (const float*)d_in[2];   // [512]
    const float* beta  = (const float*)d_in[3];   // [512]
    float*       out   = (float*)d_out;           // [32768,512]

    // workspace layout (floats): [colsum|colsq|scl|sft] then bf16 Wb^T
    float* ws     = (float*)d_ws;
    float* colsum = ws;                 // [512]
    float* colsq  = ws + NDIM;          // [512]
    float* scl    = ws + 2 * NDIM;      // [512]
    float* sft    = ws + 3 * NDIM;      // [512]
    bf16_t* WbT   = (bf16_t*)(ws + 4 * NDIM);   // [512][1024] bf16, 1 MB

    // 0) zero sum/sumsq accumulators (2*512 floats)
    zero_ws_kernel<<<(2 * NDIM) / 256, 256, 0, stream>>>(ws);

    // 1) binarize + transpose W -> bf16 Wb^T (done once, 0.5M elements)
    binarize_w_kernel<<<(KDIM * NDIM) / 256, 256, 0, stream>>>(W, WbT);

    // 2) WMMA GEMM + column stats; writes pre-activation into d_out
    dim3 grid(NDIM / 128, BDIM / 128);   // 4 x 256 workgroups
    bgemm_wmma_kernel<<<grid, 256, 0, stream>>>(x, WbT, out, colsum, colsq);

    // 3) per-column BN scale/shift
    bn_stats_kernel<<<(NDIM + 255) / 256, 256, 0, stream>>>(
        colsum, colsq, gamma, beta, scl, sft);

    // 4) in-place sign(BN(out))
    size_t nvec4 = ((size_t)BDIM * NDIM) / 4;
    bn_sign_kernel<<<(unsigned)(nvec4 / 256), 256, 0, stream>>>(out, scl, sft);
}
